// RAFTHead_52862457479819
// MI455X (gfx1250) — matrix operand order repaired and checked by reading, not compiled, not run on hardware
//
#include <hip/hip_runtime.h>
#include <hip/hip_bf16.h>

typedef __attribute__((ext_vector_type(16))) __bf16 v16bf;
typedef __attribute__((ext_vector_type(8)))  __bf16 v8bf;
typedef __attribute__((ext_vector_type(8)))  float  v8f;

#define NHW   37
#define NPIX  1369            // 37*37
#define BATCH 16
#define CIN   768
#define MROWS (BATCH * NPIX)  // 21904
#define K1    (CIN * 9)       // 6912
#define N1    512             // fh hidden (256) || wp hidden (256)
#define HIDC  256
#define K2    256
#define N2    1764            // 9*14*14
#define OPIX  518

#define BM 128
#define BN 128
#define BK 32
#define APAD 40               // LDS row stride in bf16 (80B, 16B aligned)

// ---------------------------------------------------------------- helpers ---

__device__ __forceinline__ v8bf bf8_zero() {
  v8bf z;
#pragma unroll
  for (int i = 0; i < 8; ++i) z[i] = (__bf16)0.0f;
  return z;
}

__device__ __forceinline__ v16bf pack16(v8bf lo, v8bf hi) {
  v16bf f;
#pragma unroll
  for (int i = 0; i < 8; ++i) { f[i] = lo[i]; f[8 + i] = hi[i]; }
  return f;
}

// A fragment 16x32 (MxK), ISA 7.12.2: lane<16 -> K {0..7,16..23}, lane>=16 -> {8..15,24..31}
__device__ __forceinline__ v16bf fragA(const __bf16* As, int rowbase, int lane) {
  const int r = rowbase + (lane & 15);
  const int half = lane >> 4;
  v8bf lo = *(const v8bf*)(As + r * APAD + half * 8);
  v8bf hi = *(const v8bf*)(As + r * APAD + 16 + half * 8);
  return pack16(lo, hi);
}

// B fragment 32x16 (KxN), stored N-major/K-contiguous in LDS:
// lane = N index; lanes 0-15 hold K {0..7,16..23}, lanes 16-31 hold K {8..15,24..31}
__device__ __forceinline__ v16bf fragB(const __bf16* Bs, int colbase, int lane) {
  const int n = colbase + (lane & 15);
  const int kb = (lane >> 4) * 8;
  v8bf lo = *(const v8bf*)(Bs + n * APAD + kb);
  v8bf hi = *(const v8bf*)(Bs + n * APAD + kb + 16);
  return pack16(lo, hi);
}

#define WMMA_BF16(A, Bv, C) \
  __builtin_amdgcn_wmma_f32_16x16x32_bf16(false, (A), false, (Bv), (short)0, (C), false, false)

// ------------------------------------------------------------ pack kernels --

__global__ void __launch_bounds__(256)
cast_f32_to_bf16(const float* __restrict__ src, __bf16* __restrict__ dst, int n) {
  int i = blockIdx.x * blockDim.x + threadIdx.x;
  if (i < n) dst[i] = (__bf16)src[i];
}

// w1cat[n][tap*768+c] = (n<256 ? fh_w1 : wp_w1)[n'][c][dy][dx], tap = dy*3+dx
__global__ void __launch_bounds__(256)
pack_w1(const float* __restrict__ fh, const float* __restrict__ wp,
        __bf16* __restrict__ w1cat) {
  int idx = blockIdx.x * blockDim.x + threadIdx.x;
  if (idx >= N1 * K1) return;
  int n = idx / K1, k = idx - n * K1;
  int tap = k / CIN, c = k - tap * CIN;
  const float* src = (n < HIDC) ? fh : wp;
  int nn = (n < HIDC) ? n : n - HIDC;
  w1cat[idx] = (__bf16)src[(size_t)nn * K1 + c * 9 + tap];
}

__global__ void __launch_bounds__(256)
pack_b1(const float* __restrict__ fh_b, const float* __restrict__ wp_b,
        float* __restrict__ b1cat) {
  int i = blockIdx.x * blockDim.x + threadIdx.x;
  if (i < N1) b1cat[i] = (i < HIDC) ? fh_b[i] : wp_b[i - HIDC];
}

// -------------------- GEMM1: fused 3x3 conv 768->512 (implicit im2col) ------
// hid[m][n] = ReLU( sum_k A_im2col[m][k] * w1cat[n][k] + b1cat[n] ), bf16 out
// 128x128 block tile, 8 waves (4x2), 32x64 per wave (8 WMMA / K-step),
// double-buffered LDS, one barrier per K-step.

__global__ void __launch_bounds__(256)
gemm1_im2col_wmma(const __bf16* __restrict__ tok,    // [B][37][37][768]
                  const __bf16* __restrict__ w1cat,  // [512][6912]
                  const float*  __restrict__ b1cat,  // [512]
                  __bf16* __restrict__ hid)          // [MROWS][512]
{
  __shared__ __align__(16) __bf16 Asm[2][BM * APAD];
  __shared__ __align__(16) __bf16 Bsm[2][BN * APAD];
  const int tid = threadIdx.x;
  const int lane = tid & 31;
  const int wid = tid >> 5;
  const int wm = wid & 3, wn = wid >> 2;     // 4x2 wave grid, 32x64 per wave
  const int m0 = blockIdx.x * BM;
  const int n0 = blockIdx.y * BN;

  // A loader: 2 threads per tile row, 16 bf16 each (implicit im2col)
  const int arow = tid >> 1;
  const int akh  = (tid & 1) * 16;
  const int m = m0 + arow;
  const bool mval = m < MROWS;
  int bb = 0, hh = 0, ww = 0;
  if (mval) { bb = m / NPIX; int rem = m - bb * NPIX; hh = rem / NHW; ww = rem - hh * NHW; }

  // B loader: 2 threads per tile row, 16 bf16 each
  const int brow = tid >> 1;
  const int bkh  = (tid & 1) * 16;
  const __bf16* bsrc = w1cat + (size_t)(n0 + brow) * K1 + bkh;

  auto load_a = [&](int k0, v8bf& lo, v8bf& hi) {
    int k = k0 + akh;
    int tap = k / CIN;                 // 32-wide chunk never straddles a tap
    int c = k - tap * CIN;
    int iy = hh + tap / 3 - 1;
    int ix = ww + (tap - (tap / 3) * 3) - 1;
    lo = bf8_zero(); hi = lo;
    if (mval && (unsigned)iy < NHW && (unsigned)ix < NHW) {
      const v8bf* s = (const v8bf*)(tok + ((size_t)(bb * NPIX + iy * NHW + ix)) * CIN + c);
      lo = s[0]; hi = s[1];
    }
  };
  auto store_tile = [&](int buf, v8bf alo, v8bf ahi, v8bf blo, v8bf bhi) {
    *(v8bf*)(Asm[buf] + arow * APAD + akh)     = alo;
    *(v8bf*)(Asm[buf] + arow * APAD + akh + 8) = ahi;
    *(v8bf*)(Bsm[buf] + brow * APAD + bkh)     = blo;
    *(v8bf*)(Bsm[buf] + brow * APAD + bkh + 8) = bhi;
  };

  v8f acc[2][4] = {};

  // prologue: stage K-chunk 0 into buffer 0
  {
    v8bf alo, ahi;
    load_a(0, alo, ahi);
    v8bf blo = *(const v8bf*)(bsrc);
    v8bf bhi = *(const v8bf*)(bsrc + 8);
    store_tile(0, alo, ahi, blo, bhi);
  }
  __syncthreads();

  const int KT = K1 / BK;              // 216
  for (int kt = 0; kt < KT; ++kt) {
    const int cur = kt & 1, nxt = cur ^ 1;
    v8bf alo, ahi, blo, bhi;
    const bool have_next = (kt + 1) < KT;
    if (have_next) {
      const int kn = (kt + 1) * BK;
      load_a(kn, alo, ahi);
      blo = *(const v8bf*)(bsrc + kn);
      bhi = *(const v8bf*)(bsrc + kn + 8);
      if ((kt + 2) < KT)               // pre-warm L2 two steps ahead
        __builtin_prefetch(bsrc + (kt + 2) * BK, 0, 1);
    }

    // compute on current buffer: 8 WMMAs per wave
    const __bf16* Ab = Asm[cur];
    const __bf16* Bb = Bsm[cur];
    v16bf a0 = fragA(Ab, wm * 32,      lane);
    v16bf a1 = fragA(Ab, wm * 32 + 16, lane);
    v16bf bv[4];
#pragma unroll
    for (int j = 0; j < 4; ++j) bv[j] = fragB(Bb, wn * 64 + j * 16, lane);
#pragma unroll
    for (int j = 0; j < 4; ++j) {
      acc[0][j] = WMMA_BF16(a0, bv[j], acc[0][j]);
      acc[1][j] = WMMA_BF16(a1, bv[j], acc[1][j]);
    }

    if (have_next) store_tile(nxt, alo, ahi, blo, bhi);
    __syncthreads();
  }

  // epilogue: bias + ReLU -> bf16 (C layout: row = base + (lane>=16)*8 + v, col = lane&15)
#pragma unroll
  for (int tn = 0; tn < 4; ++tn) {
    int col = n0 + wn * 64 + tn * 16 + (lane & 15);
    float bia = b1cat[col];
#pragma unroll
    for (int tm = 0; tm < 2; ++tm) {
      int rb = m0 + wm * 32 + tm * 16 + (lane >> 4) * 8;
      v8f a = acc[tm][tn];
#pragma unroll
      for (int v = 0; v < 8; ++v) {
        int r = rb + v;
        if (r < MROWS) {
          float x = a[v] + bia;
          x = x > 0.0f ? x : 0.0f;
          hid[(size_t)r * N1 + col] = (__bf16)x;
        }
      }
    }
  }
}

// ------------------- GEMM2: 1x1 conv 256->1764 (weight head), WMMA ---------

__global__ void __launch_bounds__(256)
gemm2_wmma(const __bf16* __restrict__ A, int lda,    // hid wp-half [M][lda]
           const __bf16* __restrict__ Bt,            // [N][K] K-contiguous
           const float*  __restrict__ bias,
           __bf16* __restrict__ Co, int ldc,
           int Mm, int Nn, int Kk)
{
  __shared__ __align__(16) __bf16 Asm[2][BM * APAD];
  __shared__ __align__(16) __bf16 Bsm[2][BN * APAD];
  const int tid = threadIdx.x;
  const int lane = tid & 31;
  const int wid = tid >> 5;
  const int wm = wid & 3, wn = wid >> 2;
  const int m0 = blockIdx.x * BM;
  const int n0 = blockIdx.y * BN;

  const int arow = tid >> 1;
  const int akh  = (tid & 1) * 16;
  const int m = m0 + arow;
  const bool mval = m < Mm;
  const __bf16* asrc = A + (size_t)m * lda + akh;

  const int brow = tid >> 1;
  const int bkh  = (tid & 1) * 16;
  const bool nval = (n0 + brow) < Nn;
  const __bf16* bsrc = Bt + (size_t)(n0 + brow) * Kk + bkh;

  auto load_pair = [&](int k0, v8bf& alo, v8bf& ahi, v8bf& blo, v8bf& bhi) {
    alo = bf8_zero(); ahi = alo; blo = alo; bhi = alo;
    if (mval) { const v8bf* s = (const v8bf*)(asrc + k0); alo = s[0]; ahi = s[1]; }
    if (nval) { const v8bf* s = (const v8bf*)(bsrc + k0); blo = s[0]; bhi = s[1]; }
  };
  auto store_tile = [&](int buf, v8bf alo, v8bf ahi, v8bf blo, v8bf bhi) {
    *(v8bf*)(Asm[buf] + arow * APAD + akh)     = alo;
    *(v8bf*)(Asm[buf] + arow * APAD + akh + 8) = ahi;
    *(v8bf*)(Bsm[buf] + brow * APAD + bkh)     = blo;
    *(v8bf*)(Bsm[buf] + brow * APAD + bkh + 8) = bhi;
  };

  v8f acc[2][4] = {};

  {
    v8bf alo, ahi, blo, bhi;
    load_pair(0, alo, ahi, blo, bhi);
    store_tile(0, alo, ahi, blo, bhi);
  }
  __syncthreads();

  const int KT = Kk / BK;
  for (int kt = 0; kt < KT; ++kt) {
    const int cur = kt & 1, nxt = cur ^ 1;
    v8bf alo, ahi, blo, bhi;
    const bool have_next = (kt + 1) < KT;
    if (have_next) {
      load_pair((kt + 1) * BK, alo, ahi, blo, bhi);
      if ((kt + 2) < KT) {
        __builtin_prefetch(asrc + (kt + 2) * BK, 0, 1);
        __builtin_prefetch(bsrc + (kt + 2) * BK, 0, 1);
      }
    }

    const __bf16* Ab = Asm[cur];
    const __bf16* Bb = Bsm[cur];
    v16bf a0 = fragA(Ab, wm * 32,      lane);
    v16bf a1 = fragA(Ab, wm * 32 + 16, lane);
    v16bf bv[4];
#pragma unroll
    for (int j = 0; j < 4; ++j) bv[j] = fragB(Bb, wn * 64 + j * 16, lane);
#pragma unroll
    for (int j = 0; j < 4; ++j) {
      acc[0][j] = WMMA_BF16(a0, bv[j], acc[0][j]);
      acc[1][j] = WMMA_BF16(a1, bv[j], acc[1][j]);
    }

    if (have_next) store_tile(nxt, alo, ahi, blo, bhi);
    __syncthreads();
  }

#pragma unroll
  for (int tn = 0; tn < 4; ++tn) {
    int col = n0 + wn * 64 + tn * 16 + (lane & 15);
    if (col >= Nn) continue;
    float bia = bias[col];
#pragma unroll
    for (int tm = 0; tm < 2; ++tm) {
      int rb = m0 + wm * 32 + tm * 16 + (lane >> 4) * 8;
      v8f a = acc[tm][tn];
#pragma unroll
      for (int v = 0; v < 8; ++v) {
        int r = rb + v;
        if (r < Mm) Co[(size_t)r * ldc + col] = (__bf16)(a[v] + bia);
      }
    }
  }
}

// ----------- flow head conv2: 3x3, 256 -> 3 (tiny: 0.3 GFLOP, direct) ------

__global__ void __launch_bounds__(256)
flow_conv2(const __bf16* __restrict__ hid,   // fh half = cols [0,256) of [M][512]
           const float* __restrict__ w2,     // (3,256,3,3)
           const float* __restrict__ b2, float* __restrict__ flw) {
  int idx = blockIdx.x * blockDim.x + threadIdx.x;
  if (idx >= MROWS * 3) return;
  int m = idx / 3, c3 = idx - m * 3;
  int bb = m / NPIX; int rem = m - bb * NPIX;
  int hh = rem / NHW, ww = rem - (rem / NHW) * NHW;
  float acc = b2[c3];
  for (int tap = 0; tap < 9; ++tap) {
    int iy = hh + tap / 3 - 1, ix = ww + tap % 3 - 1;
    if (iy < 0 || iy >= NHW || ix < 0 || ix >= NHW) continue;
    const __bf16* hp = hid + (size_t)(bb * NPIX + iy * NHW + ix) * N1;
    const float* wp = w2 + c3 * (HIDC * 9) + tap;
#pragma unroll 8
    for (int ch = 0; ch < HIDC; ++ch) acc += (float)hp[ch] * wp[ch * 9];
  }
  flw[m * 3 + c3] = acc;
}

// --------- fused softmax(9) + convex upsample + sigmoid, 1 block/(b,h,w) ----

__global__ void __launch_bounds__(256)
upsample_combine(const __bf16* __restrict__ wraw,  // [M][1764], col = k*196+p*14+q
                 const float* __restrict__ flw,    // [M][3]
                 float* __restrict__ out) {
  const int m = blockIdx.x;
  const int bb = m / NPIX; const int rem = m - bb * NPIX;
  const int hh = rem / NHW, ww = rem - (rem / NHW) * NHW;
  __shared__ float nb[9][3];                        // 14 * flow at 3x3 neighbors
  const int t = threadIdx.x;
  if (t < 27) {
    int k = t / 3, c = t - (t / 3) * 3;
    int iy = hh + k / 3 - 1, ix = ww + k % 3 - 1;
    float v = 0.0f;
    if (iy >= 0 && iy < NHW && ix >= 0 && ix < NHW)
      v = 14.0f * flw[(bb * NPIX + iy * NHW + ix) * 3 + c];
    nb[k][c] = v;
  }
  __syncthreads();
  if (t < 196) {
    const __bf16* wr = wraw + (size_t)m * N2 + t;
    float e[9]; float mx = -3.0e38f;
#pragma unroll
    for (int k = 0; k < 9; ++k) { e[k] = (float)wr[k * 196]; mx = fmaxf(mx, e[k]); }
    float s = 0.0f;
#pragma unroll
    for (int k = 0; k < 9; ++k) { e[k] = __expf(e[k] - mx); s += e[k]; }
    const float inv = 1.0f / s;
    float u0 = 0.0f, u1 = 0.0f, u2 = 0.0f;
#pragma unroll
    for (int k = 0; k < 9; ++k) {
      float wk = e[k] * inv;
      u0 += wk * nb[k][0]; u1 += wk * nb[k][1]; u2 += wk * nb[k][2];
    }
    int p = t / 14, q = t - p * 14;
    size_t pix = (size_t)(hh * 14 + p) * OPIX + (ww * 14 + q);
    const size_t plane = (size_t)OPIX * OPIX;
    out[(size_t)(bb * 2 + 0) * plane + pix] = u0;
    out[(size_t)(bb * 2 + 1) * plane + pix] = u1;
    out[(size_t)BATCH * 2 * plane + (size_t)bb * plane + pix] = 1.0f / (1.0f + __expf(-u2));
  }
}

// -------------------------------------------------------------- launcher ---

extern "C" void kernel_launch(void* const* d_in, const int* in_sizes, int n_in,
                              void* d_out, int out_size, void* d_ws, size_t ws_size,
                              hipStream_t stream) {
  (void)in_sizes; (void)n_in; (void)out_size; (void)ws_size;
  const float* tokens = (const float*)d_in[0];
  const float* fh_w1  = (const float*)d_in[1];
  const float* fh_b1  = (const float*)d_in[2];
  const float* fh_w2  = (const float*)d_in[3];
  const float* fh_b2  = (const float*)d_in[4];
  const float* wp_w1  = (const float*)d_in[5];
  const float* wp_b1  = (const float*)d_in[6];
  const float* wp_w2  = (const float*)d_in[7];
  const float* wp_b2  = (const float*)d_in[8];

  char* ws = (char*)d_ws;
  size_t off = 0;
  auto alloc = [&](size_t bytes) {
    void* p = ws + off;
    off = (off + bytes + 255) & ~(size_t)255;
    return p;
  };
  __bf16* tokbf = (__bf16*)alloc((size_t)MROWS * CIN * 2);   // 33.6 MB
  __bf16* w1cat = (__bf16*)alloc((size_t)N1 * K1 * 2);       //  7.1 MB
  float*  b1cat = (float*)alloc((size_t)N1 * 4);
  __bf16* hid   = (__bf16*)alloc((size_t)MROWS * N1 * 2);    // 22.4 MB
  __bf16* w2bf  = (__bf16*)alloc((size_t)N2 * K2 * 2);       //  0.9 MB
  __bf16* wraw  = (__bf16*)alloc((size_t)MROWS * N2 * 2);    // 77.3 MB
  float*  flw   = (float*)alloc((size_t)MROWS * 3 * 4);      //  0.3 MB

  const int ntok = MROWS * CIN;
  cast_f32_to_bf16<<<(ntok + 255) / 256, 256, 0, stream>>>(tokens, tokbf, ntok);
  pack_w1<<<(N1 * K1 + 255) / 256, 256, 0, stream>>>(fh_w1, wp_w1, w1cat);
  pack_b1<<<2, 256, 0, stream>>>(fh_b1, wp_b1, b1cat);
  const int nw2 = N2 * K2;
  cast_f32_to_bf16<<<(nw2 + 255) / 256, 256, 0, stream>>>(wp_w2, w2bf, nw2);

  dim3 g1((MROWS + BM - 1) / BM, N1 / BN);                   // 172 x 4
  gemm1_im2col_wmma<<<g1, 256, 0, stream>>>(tokbf, w1cat, b1cat, hid);

  dim3 g2((MROWS + BM - 1) / BM, (N2 + BN - 1) / BN);        // 172 x 14
  gemm2_wmma<<<g2, 256, 0, stream>>>(hid + HIDC, N1, w2bf, wp_b2, wraw, N2,
                                     MROWS, N2, K2);

  flow_conv2<<<(MROWS * 3 + 255) / 256, 256, 0, stream>>>(hid, fh_w2, fh_b2, flw);

  upsample_combine<<<MROWS, 256, 0, stream>>>(wraw, flw, (float*)d_out);
}
